// MLSTMBlock_14078902796641
// MI455X (gfx1250) — matrix operand order repaired and checked by reading, not compiled, not run on hardware
//
#include <hip/hip_runtime.h>
#include <hip/hip_bf16.h>
#include <cstdint>

#define B_      32
#define HID_    2048
#define INNER_  4096
#define NHEADS_ 8
#define HEAD_   512
#define KCONV_  4
#define CAPV    30.0f
#define EPS_    1e-6f
#define GN_EPS_ 1e-5f
#define INV_SQRT_HEAD 0.04419417382415922f  // 1/sqrt(512)

typedef __attribute__((ext_vector_type(16))) __bf16 v16bf;
typedef __attribute__((ext_vector_type(8)))  float  v8f;

// ---------------------------------------------------------------- RMSNorm ----
__global__ void rmsnorm_kernel(const float* __restrict__ x,
                               const float* __restrict__ w,
                               float* __restrict__ xn) {
  __shared__ float red[256];
  const int b = blockIdx.x, t = threadIdx.x;
  const float* xr = x + (size_t)b * HID_;
  float s = 0.f;
  for (int i = t; i < HID_; i += 256) { float v = xr[i]; s += v * v; }
  red[t] = s; __syncthreads();
  for (int off = 128; off > 0; off >>= 1) {
    if (t < off) red[t] += red[t + off];
    __syncthreads();
  }
  const float rs = rsqrtf(red[0] / (float)HID_ + EPS_);
  for (int i = t; i < HID_; i += 256)
    xn[(size_t)b * HID_ + i] = xr[i] * rs * w[i];
}

// -------------------------------------------------- WMMA GEMM: Y = X * W^T ---
// X: M x K row-major (fp32), W: N x K row-major (fp32), Y: M x N.
// One wave computes one 16x16 tile; K consumed 32 at a time with
// v_wmma_f32_16x16x32_bf16. Optional addend (skip connection) in epilogue.
__device__ __forceinline__ void pack8(v16bf& dst, int base, float4 u, float4 v) {
  dst[base + 0] = (__bf16)u.x; dst[base + 1] = (__bf16)u.y;
  dst[base + 2] = (__bf16)u.z; dst[base + 3] = (__bf16)u.w;
  dst[base + 4] = (__bf16)v.x; dst[base + 5] = (__bf16)v.y;
  dst[base + 6] = (__bf16)v.z; dst[base + 7] = (__bf16)v.w;
}

__global__ void gemm_wmma_bf16(const float* __restrict__ X,
                               const float* __restrict__ W,
                               float* __restrict__ Y,
                               const float* __restrict__ addend,
                               int M, int N, int K) {
  const int lane = threadIdx.x & 31;
  const int wave = threadIdx.x >> 5;
  const int l16  = lane & 15;
  const int hi   = lane >> 4;
  const int n0 = (blockIdx.x * 4 + wave) * 16;
  const int m0 = blockIdx.y * 16;

  const float* ap_base = X + (size_t)(m0 + l16) * K;  // A row for this lane
  const float* bp_base = W + (size_t)(n0 + l16) * K;  // B column = W row

  v8f acc = {};
  for (int k0 = 0; k0 < K; k0 += 32) {
    // A 16x32 bf16 fragment: lane holds K = {hi*8..+7} and {hi*8+16..+23}
    const float* ap = ap_base + k0 + hi * 8;
    float4 a0 = *(const float4*)(ap + 0);
    float4 a1 = *(const float4*)(ap + 4);
    float4 a2 = *(const float4*)(ap + 16);
    float4 a3 = *(const float4*)(ap + 20);
    // B 32x16 bf16 fragment: lane holds K = {hi*8..+7} and {hi*8+16..+23}
    const float* bp = bp_base + k0 + hi * 8;
    float4 b0 = *(const float4*)(bp + 0);
    float4 b1 = *(const float4*)(bp + 4);
    float4 b2 = *(const float4*)(bp + 16);
    float4 b3 = *(const float4*)(bp + 20);

    v16bf af, bfm;
    pack8(af, 0, a0, a1);  pack8(af, 8, a2, a3);
    pack8(bfm, 0, b0, b1); pack8(bfm, 8, b2, b3);

    acc = __builtin_amdgcn_wmma_f32_16x16x32_bf16(
        /*neg_a=*/false, af, /*neg_b=*/false, bfm,
        /*c_mod=*/(short)0, acc, /*reuse_a=*/false, /*reuse_b=*/false);
  }

  const int n = n0 + l16;
#pragma unroll
  for (int j = 0; j < 8; ++j) {
    const int m = m0 + hi * 8 + j;   // C/D layout: VGPR j -> M = j + hi*8
    float v = acc[j];
    if (addend) v += addend[(size_t)m * N + n];
    Y[(size_t)m * N + n] = v;
  }
}

// ---------------------------------------------- depthwise conv (K=4) + SiLU --
__global__ void conv_kernel(const float* __restrict__ conv_state,
                            const float* __restrict__ x_mlstm,
                            const float* __restrict__ conv_w,
                            const float* __restrict__ conv_b,
                            float* __restrict__ new_conv_state,
                            float* __restrict__ xmc) {
  const int idx = blockIdx.x * blockDim.x + threadIdx.x;   // b*INNER + c
  if (idx >= B_ * INNER_) return;
  const int c = idx % INNER_;
  const float* cs = conv_state + (size_t)idx * (KCONV_ - 1);
  const float s0 = cs[0], s1 = cs[1], s2 = cs[2];
  const float x3 = x_mlstm[idx];
  const float xc = s0 * conv_w[c * KCONV_ + 0] + s1 * conv_w[c * KCONV_ + 1] +
                   s2 * conv_w[c * KCONV_ + 2] + x3 * conv_w[c * KCONV_ + 3] +
                   conv_b[c];
  float* ncs = new_conv_state + (size_t)idx * (KCONV_ - 1);
  ncs[0] = s1; ncs[1] = s2; ncs[2] = x3;
  xmc[idx] = xc / (1.f + expf(-xc));   // SiLU
}

// -------------------------------------- gates: ig/fg GEMV + stabilized gates -
__global__ void gate_kernel(const float* __restrict__ qkv,
                            const float* __restrict__ Wi, const float* __restrict__ bi,
                            const float* __restrict__ Wf, const float* __restrict__ bfv,
                            const float* __restrict__ max_state,
                            float* __restrict__ i_gate, float* __restrict__ f_gate,
                            float* __restrict__ max_new) {
  __shared__ float si[256], sf[256];
  const int bh = blockIdx.x, b = bh / NHEADS_, h = bh % NHEADS_;
  const float* qr = qkv + (size_t)b * 3 * INNER_;
  const float* wi = Wi + (size_t)h * 3 * INNER_;
  const float* wf = Wf + (size_t)h * 3 * INNER_;
  float a = 0.f, c = 0.f;
  for (int j = threadIdx.x; j < 3 * INNER_; j += 256) {
    const float q = qr[j];
    a += q * wi[j]; c += q * wf[j];
  }
  si[threadIdx.x] = a; sf[threadIdx.x] = c; __syncthreads();
  for (int off = 128; off > 0; off >>= 1) {
    if (threadIdx.x < off) { si[threadIdx.x] += si[threadIdx.x + off];
                             sf[threadIdx.x] += sf[threadIdx.x + off]; }
    __syncthreads();
  }
  if (threadIdx.x == 0) {
    const float ig = CAPV * tanhf((si[0] + bi[h]) / CAPV);
    const float fg = CAPV * tanhf((sf[0] + bfv[h]) / CAPV);
    const float lf = fminf(fg, 0.f) - log1pf(expf(-fabsf(fg)));  // log_sigmoid
    const float ms = max_state[bh];
    const float mn = fmaxf(ig, ms + lf);
    i_gate[bh]  = expf(ig - mn);
    f_gate[bh]  = expf(lf + ms - mn);
    max_new[bh] = mn;
  }
}

// -------------------------- norm-state update + qn = q.norm_new, qk = q.kh ---
__global__ void norm_kernel(const float* __restrict__ qkv,
                            const float* __restrict__ norm_state,
                            const float* __restrict__ i_gate,
                            const float* __restrict__ f_gate,
                            float* __restrict__ norm_new,
                            float* __restrict__ qn, float* __restrict__ qk) {
  __shared__ float sq[HEAD_], sk[HEAD_];
  const int bh = blockIdx.x, b = bh / NHEADS_, h = bh % NHEADS_;
  const int d = threadIdx.x;
  const float* base = qkv + (size_t)b * 3 * INNER_ + (size_t)h * HEAD_;
  const float q  = base[d];
  const float kh = base[INNER_ + d] * INV_SQRT_HEAD;
  const float f = f_gate[bh], ig = i_gate[bh];
  const float nn = f * norm_state[(size_t)bh * HEAD_ + d] + ig * kh;
  norm_new[(size_t)bh * HEAD_ + d] = nn;
  sq[d] = q * nn; sk[d] = q * kh; __syncthreads();
  for (int off = 256; off > 0; off >>= 1) {
    if (d < off) { sq[d] += sq[d + off]; sk[d] += sk[d + off]; }
    __syncthreads();
  }
  if (d == 0) { qn[bh] = sq[0]; qk[bh] = sk[0]; }
}

// ------------- fused cell update + numer (single streaming pass over cell) --
// cell_new = f*C + i*(kh ⊗ v);  numer = f*(q·C) + i*(q·kh)*v  -> one pass.
__global__ void cell_kernel(const float* __restrict__ qkv,
                            const float* __restrict__ cell_state,
                            const float* __restrict__ i_gate,
                            const float* __restrict__ f_gate,
                            const float* __restrict__ qk,
                            float* __restrict__ cell_new,
                            float* __restrict__ numer) {
  __shared__ float qs[HEAD_], ks[HEAD_];
  const int bh = blockIdx.x, b = bh / NHEADS_, h = bh % NHEADS_;
  const float* base = qkv + (size_t)b * 3 * INNER_ + (size_t)h * HEAD_;
  for (int d = threadIdx.x; d < HEAD_; d += blockDim.x) {
    qs[d] = base[d];
    ks[d] = base[INNER_ + d] * INV_SQRT_HEAD;
  }
  __syncthreads();
  const float f = f_gate[bh], ig = i_gate[bh], qkd = qk[bh];
  const int e0 = threadIdx.x * 4;                 // 128 thr * 4 = 512 cols
  const size_t cbase = (size_t)bh * HEAD_ * HEAD_;
  const float4 v4 = *(const float4*)(base + 2 * INNER_ + e0);
  float ax = 0.f, ay = 0.f, az = 0.f, aw = 0.f;
  for (int d = 0; d < HEAD_; ++d) {
    const int dp = (d + 16 < HEAD_) ? d + 16 : HEAD_ - 1;
    __builtin_prefetch(cell_state + cbase + (size_t)dp * HEAD_ + e0, 0, 0);
    const float4 c = *(const float4*)(cell_state + cbase + (size_t)d * HEAD_ + e0);
    const float qd = qs[d];
    ax += qd * c.x; ay += qd * c.y; az += qd * c.z; aw += qd * c.w;
    const float ikd = ig * ks[d];
    float4 cn;
    cn.x = f * c.x + ikd * v4.x; cn.y = f * c.y + ikd * v4.y;
    cn.z = f * c.z + ikd * v4.z; cn.w = f * c.w + ikd * v4.w;
    *(float4*)(cell_new + cbase + (size_t)d * HEAD_ + e0) = cn;
  }
  const float iv = ig * qkd;
  float4 n4;
  n4.x = f * ax + iv * v4.x; n4.y = f * ay + iv * v4.y;
  n4.z = f * az + iv * v4.z; n4.w = f * aw + iv * v4.w;
  *(float4*)(numer + (size_t)bh * HEAD_ + e0) = n4;
}

// ---------------- denom + per-head groupnorm + gated combine -> h_buf --------
__global__ void post_kernel(const float* __restrict__ numer,
                            const float* __restrict__ qn,
                            const float* __restrict__ max_new,
                            const float* __restrict__ gn_w,
                            const float* __restrict__ gn_b,
                            const float* __restrict__ skip_w,
                            const float* __restrict__ xmc,
                            const float* __restrict__ x_gate,
                            float* __restrict__ h_buf) {
  __shared__ float s1[HEAD_], s2[HEAD_];
  const int bh = blockIdx.x, b = bh / NHEADS_, h = bh % NHEADS_;
  const int d = threadIdx.x;
  const float denom = fmaxf(fabsf(qn[bh]), expf(-max_new[bh])) + EPS_;
  const float o = numer[(size_t)bh * HEAD_ + d] / denom;
  s1[d] = o; s2[d] = o * o; __syncthreads();
  for (int off = 256; off > 0; off >>= 1) {
    if (d < off) { s1[d] += s1[d + off]; s2[d] += s2[d + off]; }
    __syncthreads();
  }
  const float mu  = s1[0] / (float)HEAD_;
  const float var = s2[0] / (float)HEAD_ - mu * mu;
  const float g = (o - mu) * rsqrtf(var + GN_EPS_);
  const int c = h * HEAD_ + d;
  const float val = g * gn_w[c] + gn_b[c];
  const float xg  = x_gate[(size_t)b * INNER_ + c];
  const float sil = xg / (1.f + expf(-xg));
  h_buf[(size_t)b * INNER_ + c] =
      (val + skip_w[c] * xmc[(size_t)b * INNER_ + c]) * sil;
}

// ---------------------------------------------------------------- launcher ---
extern "C" void kernel_launch(void* const* d_in, const int* in_sizes, int n_in,
                              void* d_out, int out_size, void* d_ws, size_t ws_size,
                              hipStream_t stream) {
  const float* x          = (const float*)d_in[0];
  const float* conv_state = (const float*)d_in[1];
  const float* cell_state = (const float*)d_in[2];
  const float* norm_state = (const float*)d_in[3];
  const float* max_state  = (const float*)d_in[4];
  const float* rms_w      = (const float*)d_in[5];
  const float* Wx         = (const float*)d_in[6];
  const float* Wg         = (const float*)d_in[7];
  const float* Wqkv       = (const float*)d_in[8];
  const float* conv_w     = (const float*)d_in[9];
  const float* conv_b     = (const float*)d_in[10];
  const float* Wi         = (const float*)d_in[11];
  const float* bi         = (const float*)d_in[12];
  const float* Wf         = (const float*)d_in[13];
  const float* bfv        = (const float*)d_in[14];
  const float* gn_w       = (const float*)d_in[15];
  const float* gn_b       = (const float*)d_in[16];
  const float* Wdown      = (const float*)d_in[17];
  const float* skip_w     = (const float*)d_in[18];

  // Output layout (concatenated flat, return order):
  float* out      = (float*)d_out;
  float* y_out    = out;                 //  32*2048        = 65536
  float* conv_out = out + 65536;         //  32*4096*3      = 393216
  float* cell_out = out + 458752;        //  32*8*512*512   = 67108864
  float* norm_out = out + 67567616;      //  32*8*512       = 131072
  float* max_out  = out + 67698688;      //  32*8           = 256

  float* ws       = (float*)d_ws;
  float* xn       = ws;                  // 65536
  float* x_mlstm  = ws + 65536;          // 131072
  float* x_gate   = ws + 196608;         // 131072
  float* xmc      = ws + 327680;         // 131072
  float* qkv      = ws + 458752;         // 393216
  float* i_g      = ws + 851968;         // 256
  float* f_g      = ws + 852224;         // 256
  float* qk       = ws + 852480;         // 256
  float* qn       = ws + 852736;         // 256
  float* numer    = ws + 852992;         // 131072
  float* h_buf    = ws + 984064;         // 131072

  rmsnorm_kernel<<<B_, 256, 0, stream>>>(x, rms_w, xn);

  gemm_wmma_bf16<<<dim3(INNER_ / 64, 2), 128, 0, stream>>>(
      xn, Wx, x_mlstm, nullptr, B_, INNER_, HID_);
  gemm_wmma_bf16<<<dim3(INNER_ / 64, 2), 128, 0, stream>>>(
      xn, Wg, x_gate, nullptr, B_, INNER_, HID_);

  conv_kernel<<<(B_ * INNER_) / 256, 256, 0, stream>>>(
      conv_state, x_mlstm, conv_w, conv_b, conv_out, xmc);

  gemm_wmma_bf16<<<dim3(3 * INNER_ / 64, 2), 128, 0, stream>>>(
      xmc, Wqkv, qkv, nullptr, B_, 3 * INNER_, INNER_);

  gate_kernel<<<B_ * NHEADS_, 256, 0, stream>>>(
      qkv, Wi, bi, Wf, bfv, max_state, i_g, f_g, max_out);

  norm_kernel<<<B_ * NHEADS_, HEAD_, 0, stream>>>(
      qkv, norm_state, i_g, f_g, norm_out, qn, qk);

  cell_kernel<<<B_ * NHEADS_, 128, 0, stream>>>(
      qkv, cell_state, i_g, f_g, qk, cell_out, numer);

  post_kernel<<<B_ * NHEADS_, HEAD_, 0, stream>>>(
      numer, qn, max_out, gn_w, gn_b, skip_w, xmc, x_gate, h_buf);

  gemm_wmma_bf16<<<dim3(HID_ / 64, 2), 128, 0, stream>>>(
      h_buf, Wdown, y_out, x, B_, HID_, INNER_);
}
